// MultiHeadSelfAttention_36550171689103
// MI455X (gfx1250) — matrix operand order repaired and checked by reading, not compiled, not run on hardware
//
#include <hip/hip_runtime.h>

typedef _Float16 v8h  __attribute__((ext_vector_type(8)));
typedef _Float16 v16h __attribute__((ext_vector_type(16)));
typedef float    v8f  __attribute__((ext_vector_type(8)));

#define S_LEN  2048
#define DMODEL 1024
#define NHEAD  16
#define DK     64

#if __has_builtin(__builtin_amdgcn_exp2f)
#define EXP2F(x) __builtin_amdgcn_exp2f(x)
#else
#define EXP2F(x) exp2f(x)
#endif
#if __has_builtin(__builtin_amdgcn_rcpf)
#define RCPF(x) __builtin_amdgcn_rcpf(x)
#else
#define RCPF(x) (1.0f / (x))
#endif

#define LOG2E      1.44269504088896340f
#define QSCALE_F16 ((_Float16)(0.125f * LOG2E))     // folds 1/sqrt(dk) and ln->log2
#define FREQ_LOG2  0.207620505931280978f            // log2(10000)/64

static __device__ __forceinline__ v16h cat16(v8h a, v8h b) {
    return __builtin_shufflevector(a, b, 0,1,2,3,4,5,6,7,8,9,10,11,12,13,14,15);
}

static __device__ __forceinline__ v8f wmma_f32_16x16x32(v16h a, v16h b, v8f c) {
    // (neg_a, A, neg_b, B, c_mod, C, reuse_a, reuse_b)
    return __builtin_amdgcn_wmma_f32_16x16x32_f16(false, a, false, b, (short)0, c, false, false);
}

// ---------------------------------------------------------------------------
// Kernel 1: qkv = x @ qkv_w^T + b, fused RoPE on q/k, scatter to
//   qf, kf : [B,H,S,64]  (f16)     vt : [B,H,64,S] (f16, transposed)
// Block 256 thr (8 waves); tile 128x128; K-step 32; reg-staged double buffer.
// ---------------------------------------------------------------------------
__global__ __launch_bounds__(256)
void qkv_rope_kernel(const float* __restrict__ X, const float* __restrict__ W,
                     const float* __restrict__ bias,
                     _Float16* __restrict__ qf, _Float16* __restrict__ kf,
                     _Float16* __restrict__ vt)
{
    __shared__ _Float16 As[128][40];
    __shared__ _Float16 Bs[128][40];

    const int tid  = threadIdx.x;
    const int wid  = tid >> 5;
    const int lane = tid & 31;
    const int l15  = lane & 15;
    const int hi   = lane >> 4;
    const int m0   = blockIdx.x * 128;
    const int n0   = blockIdx.y * 128;

    v8f acc[8] = {};

    // preload k-step 0 into registers
    float aReg[16], bReg[16];
    #pragma unroll
    for (int i = 0; i < 16; ++i) {
        int idx = i * 256 + tid, row = idx >> 5, k = idx & 31;
        aReg[i] = X[(size_t)(m0 + row) * DMODEL + k];
        bReg[i] = W[(size_t)(n0 + row) * DMODEL + k];
    }

    for (int kk = 0; kk < DMODEL; kk += 32) {
        __syncthreads();                       // prev-iter LDS readers done
        #pragma unroll
        for (int i = 0; i < 16; ++i) {
            int idx = i * 256 + tid, row = idx >> 5, k = idx & 31;
            As[row][k] = (_Float16)aReg[i];
            Bs[row][k] = (_Float16)bReg[i];
        }
        __syncthreads();

        if (kk + 32 < DMODEL) {                // overlap next loads with WMMA
            #pragma unroll
            for (int i = 0; i < 16; ++i) {
                int idx = i * 256 + tid, row = idx >> 5, k = idx & 31;
                aReg[i] = X[(size_t)(m0 + row) * DMODEL + kk + 32 + k];
                bReg[i] = W[(size_t)(n0 + row) * DMODEL + kk + 32 + k];
            }
            if (kk + 64 < DMODEL) {            // gfx1250 global_prefetch_b8
                int idx = tid * 16, row = idx >> 5, k = idx & 31;
                __builtin_prefetch(&X[(size_t)(m0 + row) * DMODEL + kk + 64 + k], 0, 1);
                __builtin_prefetch(&W[(size_t)(n0 + row) * DMODEL + kk + 64 + k], 0, 1);
            }
        }

        // A fragment: lane row = wid*16+l15; K = 16*hi..+15
        v8h alo = *(const v8h*)&As[wid * 16 + l15][16 * hi];
        v8h ahi = *(const v8h*)&As[wid * 16 + l15][16 * hi + 8];
        v16h a  = cat16(alo, ahi);

        v16h bfr[8];
        #pragma unroll
        for (int nt = 0; nt < 8; ++nt) {       // batch all ds_loads
            v8h blo = *(const v8h*)&Bs[nt * 16 + l15][16 * hi];
            v8h bhi = *(const v8h*)&Bs[nt * 16 + l15][16 * hi + 8];
            bfr[nt] = cat16(blo, bhi);
        }
        #pragma unroll
        for (int nt = 0; nt < 8; ++nt)
            acc[nt] = wmma_f32_16x16x32(a, bfr[nt], acc[nt]);
    }

    // Epilogue: bias + RoPE + scatter. D layout: lane col = n, rows = r+8*hi.
    #pragma unroll
    for (int nt = 0; nt < 8; ++nt) {
        int   n    = n0 + nt * 16 + l15;       // uniform section per wave
        float bv   = bias[n];
        int   sect = n >> 10;                  // 0=q 1=k 2=v
        int   nn   = n & 1023;
        int   h    = nn >> 6, dd = nn & 63;
        float freq = EXP2F(-(float)(dd & ~1) * FREQ_LOG2);  // one v_exp_f32

        #pragma unroll
        for (int r = 0; r < 8; ++r) {
            int   m   = m0 + wid * 16 + r + 8 * hi;
            int   b   = m >> 11, s = m & (S_LEN - 1);
            float val = acc[nt][r] + bv;
            if (sect == 2) {
                vt[(((size_t)(b * NHEAD + h)) * DK + dd) * S_LEN + s] = (_Float16)val;
            } else {
                float part = __shfl_xor(val, 1, 32);   // even/odd pair in adjacent lanes
                float sn, cs;
                __sincosf((float)s * freq, &sn, &cs);
                float res = (dd & 1) ? (part * sn + val * cs)
                                     : (val * cs - part * sn);
                _Float16* dst = (sect == 0) ? qf : kf;
                dst[(((size_t)(b * NHEAD + h)) * S_LEN + s) * DK + dd] = (_Float16)res;
            }
        }
    }
}

// ---------------------------------------------------------------------------
// Kernel 2: causal flash attention, log2-domain softmax.
// One wave per 16-row Q tile; kv step 32; per-wave LDS for P transpose.
// ---------------------------------------------------------------------------
__global__ __launch_bounds__(256)
void flash_attn_kernel(const _Float16* __restrict__ qf,
                       const _Float16* __restrict__ kf,
                       const _Float16* __restrict__ vt,
                       _Float16* __restrict__ yf)
{
    __shared__ _Float16 P[8][16][40];

    const int tid  = threadIdx.x;
    const int wid  = tid >> 5;
    const int lane = tid & 31;
    const int l15  = lane & 15;
    const int hi   = lane >> 4;

    const int t     = blockIdx.x * 8 + wid;
    const int bh    = t >> 7;
    const int qt    = t & 127;
    const int qbase = qt * 16;

    const _Float16* qrow = qf + (size_t)bh * S_LEN * DK;
    const _Float16* krow = kf + (size_t)bh * S_LEN * DK;
    const _Float16* vbas = vt + (size_t)bh * DK * S_LEN;

    // Q fragments, pre-scaled by 0.125*log2e (score scale + base-2 softmax)
    v16h qfr[2];
    {
        const _Float16* qp = qrow + (size_t)(qbase + l15) * DK;
        #pragma unroll
        for (int ks = 0; ks < 2; ++ks) {
            v8h lo = *(const v8h*)(qp + ks * 32 + 16 * hi);
            v8h hb = *(const v8h*)(qp + ks * 32 + 16 * hi + 8);
            qfr[ks] = cat16(lo, hb) * QSCALE_F16;
        }
    }

    v8f   acc[4] = {};
    float mrow[8], lrow[8];
    #pragma unroll
    for (int r = 0; r < 8; ++r) { mrow[r] = -1e30f; lrow[r] = 0.0f; }

    for (int kv0 = 0; kv0 <= qbase + 15; kv0 += 32) {
        // --- scores (log2 domain): batch K-fragment loads, then 4 WMMAs ---
        v16h kfr[2][2];
        #pragma unroll
        for (int c = 0; c < 2; ++c) {
            const _Float16* kp = krow + (size_t)(kv0 + c * 16 + l15) * DK;
            #pragma unroll
            for (int ks = 0; ks < 2; ++ks) {
                v8h lo = *(const v8h*)(kp + ks * 32 + 16 * hi);
                v8h hb = *(const v8h*)(kp + ks * 32 + 16 * hi + 8);
                kfr[c][ks] = cat16(lo, hb);
            }
        }
        v8f s[2] = {};
        #pragma unroll
        for (int c = 0; c < 2; ++c)
            #pragma unroll
            for (int ks = 0; ks < 2; ++ks)
                s[c] = wmma_f32_16x16x32(qfr[ks], kfr[c][ks], s[c]);

        // --- online softmax: rows live across 16 lanes of each half ---
        float corr[8], e0[8], e1[8];
        #pragma unroll
        for (int r = 0; r < 8; ++r) {
            int   qi = qbase + r + 8 * hi;
            float a0 = s[0][r]; if (kv0 +      l15 > qi) a0 = -1e30f;
            float a1 = s[1][r]; if (kv0 + 16 + l15 > qi) a1 = -1e30f;
            float mx = fmaxf(a0, a1);
            mx = fmaxf(mx, __shfl_xor(mx, 1, 16));
            mx = fmaxf(mx, __shfl_xor(mx, 2, 16));
            mx = fmaxf(mx, __shfl_xor(mx, 4, 16));
            mx = fmaxf(mx, __shfl_xor(mx, 8, 16));
            float mnew = fmaxf(mrow[r], mx);
            corr[r] = EXP2F(mrow[r] - mnew);      // one v_exp_f32 each
            mrow[r] = mnew;
            e0[r] = EXP2F(a0 - mnew);
            e1[r] = EXP2F(a1 - mnew);
            float rs = e0[r] + e1[r];
            rs += __shfl_xor(rs, 1, 16);
            rs += __shfl_xor(rs, 2, 16);
            rs += __shfl_xor(rs, 4, 16);
            rs += __shfl_xor(rs, 8, 16);
            lrow[r] = lrow[r] * corr[r] + rs;
        }
        #pragma unroll
        for (int nt = 0; nt < 4; ++nt)
            #pragma unroll
            for (int r = 0; r < 8; ++r)
                acc[nt][r] *= corr[r];

        // --- transpose P (C-layout) -> A-fragment via per-wave LDS ---
        #pragma unroll
        for (int r = 0; r < 8; ++r) {
            P[wid][r + 8 * hi][l15]      = (_Float16)e0[r];
            P[wid][r + 8 * hi][16 + l15] = (_Float16)e1[r];
        }
        asm volatile("s_wait_dscnt 0" ::: "memory");   // same-wave LDS RAW fence
        v8h plo = *(const v8h*)&P[wid][l15][16 * hi];
        v8h phi = *(const v8h*)&P[wid][l15][16 * hi + 8];
        v16h pfr = cat16(plo, phi);

        // --- P @ V: batch Vt fragment loads, then 4 WMMAs ---
        v16h vfr[4];
        #pragma unroll
        for (int nt = 0; nt < 4; ++nt) {
            const _Float16* vp = vbas + (size_t)(nt * 16 + l15) * S_LEN + kv0 + 16 * hi;
            vfr[nt] = cat16(*(const v8h*)vp, *(const v8h*)(vp + 8));
        }
        #pragma unroll
        for (int nt = 0; nt < 4; ++nt)
            acc[nt] = wmma_f32_16x16x32(pfr, vfr[nt], acc[nt]);
    }

    // epilogue: normalize (v_rcp_f32), write y[b][s][h*64+dd] as f16
    const int b = bh >> 4, h = bh & 15;
    float rinv[8];
    #pragma unroll
    for (int r = 0; r < 8; ++r) rinv[r] = RCPF(lrow[r]);
    #pragma unroll
    for (int nt = 0; nt < 4; ++nt) {
        int dd = nt * 16 + l15;
        #pragma unroll
        for (int r = 0; r < 8; ++r) {
            int sq = qbase + r + 8 * hi;
            yf[((size_t)b * S_LEN + sq) * DMODEL + h * DK + dd] =
                (_Float16)(acc[nt][r] * rinv[r]);
        }
    }
}

// ---------------------------------------------------------------------------
// Kernel 3: out = y @ proj_w^T + proj_b  (f16 x f16 -> f32)
// ---------------------------------------------------------------------------
__global__ __launch_bounds__(256)
void proj_kernel(const _Float16* __restrict__ Y, const float* __restrict__ W,
                 const float* __restrict__ bias, float* __restrict__ out)
{
    __shared__ _Float16 As[128][40];
    __shared__ _Float16 Bs[128][40];

    const int tid  = threadIdx.x;
    const int wid  = tid >> 5;
    const int lane = tid & 31;
    const int l15  = lane & 15;
    const int hi   = lane >> 4;
    const int m0   = blockIdx.x * 128;
    const int n0   = blockIdx.y * 128;

    v8f acc[8] = {};

    _Float16 aReg[16];
    float    bReg[16];
    #pragma unroll
    for (int i = 0; i < 16; ++i) {
        int idx = i * 256 + tid, row = idx >> 5, k = idx & 31;
        aReg[i] = Y[(size_t)(m0 + row) * DMODEL + k];
        bReg[i] = W[(size_t)(n0 + row) * DMODEL + k];
    }

    for (int kk = 0; kk < DMODEL; kk += 32) {
        __syncthreads();
        #pragma unroll
        for (int i = 0; i < 16; ++i) {
            int idx = i * 256 + tid, row = idx >> 5, k = idx & 31;
            As[row][k] = aReg[i];
            Bs[row][k] = (_Float16)bReg[i];
        }
        __syncthreads();

        if (kk + 32 < DMODEL) {
            #pragma unroll
            for (int i = 0; i < 16; ++i) {
                int idx = i * 256 + tid, row = idx >> 5, k = idx & 31;
                aReg[i] = Y[(size_t)(m0 + row) * DMODEL + kk + 32 + k];
                bReg[i] = W[(size_t)(n0 + row) * DMODEL + kk + 32 + k];
            }
        }

        v8h alo = *(const v8h*)&As[wid * 16 + l15][16 * hi];
        v8h ahi = *(const v8h*)&As[wid * 16 + l15][16 * hi + 8];
        v16h a  = cat16(alo, ahi);

        v16h bfr[8];
        #pragma unroll
        for (int nt = 0; nt < 8; ++nt) {
            v8h blo = *(const v8h*)&Bs[nt * 16 + l15][16 * hi];
            v8h bhi = *(const v8h*)&Bs[nt * 16 + l15][16 * hi + 8];
            bfr[nt] = cat16(blo, bhi);
        }
        #pragma unroll
        for (int nt = 0; nt < 8; ++nt)
            acc[nt] = wmma_f32_16x16x32(a, bfr[nt], acc[nt]);
    }

    #pragma unroll
    for (int nt = 0; nt < 8; ++nt) {
        int   n  = n0 + nt * 16 + l15;
        float bv = bias[n];
        #pragma unroll
        for (int r = 0; r < 8; ++r) {
            int m = m0 + wid * 16 + r + 8 * hi;
            out[(size_t)m * DMODEL + n] = acc[nt][r] + bv;
        }
    }
}

// ---------------------------------------------------------------------------
extern "C" void kernel_launch(void* const* d_in, const int* in_sizes, int n_in,
                              void* d_out, int out_size, void* d_ws, size_t ws_size,
                              hipStream_t stream)
{
    const float* x      = (const float*)d_in[0];
    const float* qkv_w  = (const float*)d_in[1];
    const float* qkv_b  = (const float*)d_in[2];
    const float* proj_w = (const float*)d_in[3];
    const float* proj_b = (const float*)d_in[4];
    float*       out    = (float*)d_out;

    const size_t NELT = (size_t)4 * NHEAD * S_LEN * DK;   // 8,388,608 halfs each
    _Float16* qf = (_Float16*)d_ws;
    _Float16* kf = qf + NELT;
    _Float16* vt = kf + NELT;
    _Float16* yf = vt + NELT;

    qkv_rope_kernel<<<dim3(64, 24), 256, 0, stream>>>(x, qkv_w, qkv_b, qf, kf, vt);
    flash_attn_kernel<<<1024, 256, 0, stream>>>(qf, kf, vt, yf);
    proj_kernel<<<dim3(64, 8), 256, 0, stream>>>(yf, proj_w, proj_b, out);
}
